// WindowAttention_10952166605163
// MI455X (gfx1250) — compile-verified, matrix operand order checked by reading, not run on hardware
//
#include <hip/hip_runtime.h>
#include <hip/hip_bf16.h>

// ---------------------------------------------------------------------------
// Fused Swin window-attention for gfx1250 (MI455X), bf16 WMMA + f32 accum.
//   B_=2048 windows, N=49 (pad 64), C=256, H=8 heads, d=32, nW=64.
// One workgroup (8 wave32) per window; wave==head in the attention phase.
// ---------------------------------------------------------------------------

typedef __attribute__((ext_vector_type(16))) __bf16 v16bf;
typedef __attribute__((ext_vector_type(8)))  float  v8f;

union ABfrag {
    v16bf v;
    unsigned short h[16];
    unsigned int   u[8];
};

__device__ __forceinline__ unsigned short f32_to_bf16(float f) {
    unsigned int u = __float_as_uint(f);
    u += 0x7FFFu + ((u >> 16) & 1u);   // round-to-nearest-even
    return (unsigned short)(u >> 16);
}

__device__ __forceinline__ v8f wmma_bf16(v16bf a, v16bf b, v8f c) {
    return __builtin_amdgcn_wmma_f32_16x16x32_bf16(
        /*neg_a=*/false, a, /*neg_b=*/false, b,
        /*c_mod=*/(short)0, c, /*reuse_a=*/false, /*reuse_b=*/false);
}

// A-operand (16x32 bf16, ISA 7.12.2 layout) gathered from a row-major LDS
// tile.  Lane l holds row m = mb + (l&15); K pairs:
//   vr 0..3 -> K = khalf + 2*vr,  vr 4..7 -> K = 16 + khalf + 2*(vr-4)
// with khalf = 8*(l>=16).  Identical layout serves k-transpose B operand.
__device__ __forceinline__ v16bf load_afrag_lds(const unsigned short* base,
                                                int stride, int mb, int kb,
                                                int lane) {
    ABfrag f;
    const unsigned short* row = base + (mb + (lane & 15)) * stride + kb;
    const int khalf = (lane >> 4) << 3;
#pragma unroll
    for (int vr = 0; vr < 8; ++vr) {
        int k = ((vr >> 2) << 4) + khalf + ((vr & 3) << 1);
        f.h[2 * vr]     = row[k];
        f.h[2 * vr + 1] = row[k + 1];
    }
    return f.v;
}

// B-operand (32x16 bf16) for P@V: element (k=n, col) lives at LDS row n.
__device__ __forceinline__ v16bf load_bvfrag_lds(const unsigned short* base,
                                                 int stride, int nb, int lane) {
    ABfrag f;
    const int c = lane & 15;
    const int khalf = (lane >> 4) << 3;
#pragma unroll
    for (int vr = 0; vr < 8; ++vr) {
        int n = nb + ((vr >> 2) << 4) + khalf + ((vr & 3) << 1);
        f.h[2 * vr]     = base[n * stride + c];
        f.h[2 * vr + 1] = base[(n + 1) * stride + c];
    }
    return f.v;
}

// B-operand from pre-swizzled bf16 weights: 32 contiguous bytes per lane.
__device__ __forceinline__ v16bf load_bfrag_gl(const unsigned short* p) {
    ABfrag f;
    const uint4* q = (const uint4*)p;
    uint4 a0 = q[0];
    uint4 a1 = q[1];
    f.u[0] = a0.x; f.u[1] = a0.y; f.u[2] = a0.z; f.u[3] = a0.w;
    f.u[4] = a1.x; f.u[5] = a1.y; f.u[6] = a1.z; f.u[7] = a1.w;
    return f.v;
}

// ---------------------------------------------------------------------------
// Prep: f32 -> bf16 weights, pre-swizzled into WMMA B-operand order.
// Swizzled flat index = ((nt*KB + kb)*32 + lane)*16 + e  with
//   n = nt*16 + (lane&15)
//   k = kb*32 + 16*(e>>3) + 8*(lane>>4) + 2*((e>>1)&3) + (e&1)
// qkv_w: 48 n-tiles x 8 k-blocks; proj_w: 16 x 8.
// ---------------------------------------------------------------------------
__global__ void prep_weights_kernel(const float* __restrict__ qkv_w,
                                    const float* __restrict__ proj_w,
                                    unsigned short* __restrict__ ws) {
    const int QKV_ELEMS = 48 * 8 * 512;   // 196608
    const int PRJ_ELEMS = 16 * 8 * 512;   //  65536
    int idx = blockIdx.x * blockDim.x + threadIdx.x;
    if (idx < QKV_ELEMS) {
        int e = idx & 15, lane = (idx >> 4) & 31, r = idx >> 9;
        int kb = r & 7, nt = r >> 3;
        int n = nt * 16 + (lane & 15);
        int vr = e >> 1, p = e & 1;
        int k = kb * 32 + ((vr >> 2) << 4) + ((lane >> 4) << 3) + ((vr & 3) << 1) + p;
        ws[idx] = f32_to_bf16(qkv_w[k * 768 + n]);
    } else if (idx < QKV_ELEMS + PRJ_ELEMS) {
        int j = idx - QKV_ELEMS;
        int e = j & 15, lane = (j >> 4) & 31, r = j >> 9;
        int kb = r & 7, nt = r >> 3;
        int n = nt * 16 + (lane & 15);
        int vr = e >> 1, p = e & 1;
        int k = kb * 32 + ((vr >> 2) << 4) + ((lane >> 4) << 3) + ((vr & 3) << 1) + p;
        ws[QKV_ELEMS + j] = f32_to_bf16(proj_w[k * 256 + n]);
    }
}

// ---------------------------------------------------------------------------
// Fused window attention.
// ---------------------------------------------------------------------------
#define QSTRIDE 776   // 768 + 8 bf16 pad  (388 dwords % 64 == 4 -> no conflicts)
#define XSTRIDE 264   // 256 + 8
#define PSTRIDE 72    //  64 + 8
#define XO_SHORTS (64 * XSTRIDE)            // 16896: x staging / attended out
#define P_SHORTS  (8 * 64 * PSTRIDE)        // 36864: per-wave softmax probs

__global__ __launch_bounds__(256, 1)
void win_attn_kernel(const float* __restrict__ x,
                     const float* __restrict__ mask,
                     const float* __restrict__ qkv_b,
                     const float* __restrict__ proj_b,
                     const float* __restrict__ bias_table,
                     const unsigned short* __restrict__ wq,   // swizzled qkv_w bf16
                     const unsigned short* __restrict__ wp,   // swizzled proj_w bf16
                     float* __restrict__ out) {
    __shared__ unsigned short qkvs[64 * QSTRIDE];         // 99,328 B
    __shared__ unsigned short xpo[XO_SHORTS + P_SHORTS];  // 107,520 B

    const int b    = blockIdx.x;
    const int tid  = threadIdx.x;
    const int wave = tid >> 5;
    const int lane = tid & 31;

    // ---- Phase 1: stage x window as bf16 (rows 49..63 zero) ----------------
    const float* xw = x + (size_t)b * 49 * 256;
    for (int i = tid; i < 64 * 256; i += 256) {
        int m = i >> 8, c = i & 255;
        float v = (m < 49) ? xw[m * 256 + c] : 0.f;
        xpo[m * XSTRIDE + c] = f32_to_bf16(v);
    }
    __syncthreads();

    // ---- Phase 2: qkv = x @ qkv_w + qkv_b  (q pre-scaled by d^-0.5) --------
    {
        const float scale = 0.17677669529663687f;   // 32^-0.5
#pragma unroll 1
        for (int t = 0; t < 6; ++t) {
            int nt  = wave * 6 + t;
            int col = nt * 16 + (lane & 15);
            float bias = qkv_b[col];
            float scl  = (col < 256) ? scale : 1.0f;
#pragma unroll 1
            for (int mt = 0; mt < 4; ++mt) {
                v8f acc = {};
#pragma unroll
                for (int kb = 0; kb < 8; ++kb) {
                    v16bf a  = load_afrag_lds(xpo, XSTRIDE, mt * 16, kb * 32, lane);
                    v16bf bw = load_bfrag_gl(wq + ((nt * 8 + kb) * 32 + lane) * 16);
                    acc = wmma_bf16(a, bw, acc);
                }
                int mbase = mt * 16 + ((lane >> 4) << 3);
#pragma unroll
                for (int vr = 0; vr < 8; ++vr)
                    qkvs[(mbase + vr) * QSTRIDE + col] =
                        f32_to_bf16((acc[vr] + bias) * scl);
            }
        }
    }
    __syncthreads();

    // ---- Phase 3: per-head attention (wave == head) ------------------------
    {
        const int h = wave;
        const unsigned short* qbase = qkvs + h * 32;          // q cols
        const unsigned short* kbase = qkvs + 256 + h * 32;    // k cols
        const unsigned short* vbase = qkvs + 512 + h * 32;    // v cols
        const float* maskw = mask + (size_t)(b & 63) * 49 * 49;
        unsigned short* pb = xpo + XO_SHORTS + wave * 64 * PSTRIDE;

#pragma unroll 1
        for (int mt = 0; mt < 4; ++mt) {
            // S = (q*scale) @ k^T : K = 32, one WMMA per n-tile.
            v16bf aq = load_afrag_lds(qbase, QSTRIDE, mt * 16, 0, lane);
            v8f s[4];
#pragma unroll
            for (int nt = 0; nt < 4; ++nt) {
                v8f acc = {};
                v16bf bk = load_afrag_lds(kbase, QSTRIDE, nt * 16, 0, lane);
                s[nt] = wmma_bf16(aq, bk, acc);
            }
            // + relative-position bias + window mask; pad cols -> -inf.
            const int mbase = mt * 16 + ((lane >> 4) << 3);
#pragma unroll
            for (int nt = 0; nt < 4; ++nt) {
                int n = nt * 16 + (lane & 15);
#pragma unroll
                for (int vr = 0; vr < 8; ++vr) {
                    int m  = mbase + vr;
                    float v = s[nt][vr];
                    if (n < 49) {
                        int mm = (m < 49) ? m : 0;
                        int d0 = mm / 7 - n / 7 + 6;
                        int d1 = mm % 7 - n % 7 + 6;
                        v += bias_table[(d0 * 13 + d1) * 8 + h] + maskw[mm * 49 + n];
                    } else {
                        v = -1e30f;
                    }
                    s[nt][vr] = v;
                }
            }
            // Row softmax: each row lives in one 16-lane half (D layout).
            float rm[8], rs[8];
#pragma unroll
            for (int vr = 0; vr < 8; ++vr) {
                float m0 = s[0][vr];
#pragma unroll
                for (int nt = 1; nt < 4; ++nt) m0 = fmaxf(m0, s[nt][vr]);
                for (int off = 1; off < 16; off <<= 1)
                    m0 = fmaxf(m0, __shfl_xor(m0, off, 16));
                rm[vr] = m0;
            }
#pragma unroll
            for (int vr = 0; vr < 8; ++vr) {
                float sum = 0.f;
#pragma unroll
                for (int nt = 0; nt < 4; ++nt) {
                    float e = __expf(s[nt][vr] - rm[vr]);
                    s[nt][vr] = e;
                    sum += e;
                }
                for (int off = 1; off < 16; off <<= 1)
                    sum += __shfl_xor(sum, off, 16);
                rs[vr] = 1.0f / sum;
            }
            // Normalized P -> bf16 per-wave LDS buffer.
#pragma unroll
            for (int nt = 0; nt < 4; ++nt) {
                int n = nt * 16 + (lane & 15);
#pragma unroll
                for (int vr = 0; vr < 8; ++vr)
                    pb[(mbase + vr) * PSTRIDE + n] =
                        f32_to_bf16(s[nt][vr] * rs[vr]);
            }
        }

        // O = P @ V  (K = 64 over tokens, 2 d-tiles of 16)
#pragma unroll 1
        for (int mt = 0; mt < 4; ++mt) {
#pragma unroll
            for (int dt = 0; dt < 2; ++dt) {
                v8f acc = {};
#pragma unroll
                for (int kb = 0; kb < 2; ++kb) {
                    v16bf ap = load_afrag_lds(pb, PSTRIDE, mt * 16, kb * 32, lane);
                    v16bf bv = load_bvfrag_lds(vbase + dt * 16, QSTRIDE, kb * 32, lane);
                    acc = wmma_bf16(ap, bv, acc);
                }
                int col   = h * 32 + dt * 16 + (lane & 15);
                int mbase = mt * 16 + ((lane >> 4) << 3);
#pragma unroll
                for (int vr = 0; vr < 8; ++vr)
                    xpo[(mbase + vr) * XSTRIDE + col] = f32_to_bf16(acc[vr]);
            }
        }
    }
    __syncthreads();

    // ---- Phase 4: out = O @ proj_w + proj_b --------------------------------
    {
#pragma unroll 1
        for (int t = 0; t < 2; ++t) {
            int nt  = wave * 2 + t;
            int col = nt * 16 + (lane & 15);
            float bias = proj_b[col];
#pragma unroll 1
            for (int mt = 0; mt < 4; ++mt) {
                v8f acc = {};
#pragma unroll
                for (int kb = 0; kb < 8; ++kb) {
                    v16bf a  = load_afrag_lds(xpo, XSTRIDE, mt * 16, kb * 32, lane);
                    v16bf bw = load_bfrag_gl(wp + ((nt * 8 + kb) * 32 + lane) * 16);
                    acc = wmma_bf16(a, bw, acc);
                }
                int mbase = mt * 16 + ((lane >> 4) << 3);
#pragma unroll
                for (int vr = 0; vr < 8; ++vr) {
                    int m = mbase + vr;
                    if (m < 49)
                        out[((size_t)b * 49 + m) * 256 + col] = acc[vr] + bias;
                }
            }
        }
    }
}

// ---------------------------------------------------------------------------
extern "C" void kernel_launch(void* const* d_in, const int* in_sizes, int n_in,
                              void* d_out, int out_size, void* d_ws, size_t ws_size,
                              hipStream_t stream) {
    const float* x          = (const float*)d_in[0];
    const float* mask       = (const float*)d_in[1];
    const float* qkv_w      = (const float*)d_in[2];
    const float* qkv_b      = (const float*)d_in[3];
    const float* proj_w     = (const float*)d_in[4];
    const float* proj_b     = (const float*)d_in[5];
    const float* bias_table = (const float*)d_in[6];
    float* out = (float*)d_out;

    unsigned short* ws = (unsigned short*)d_ws;
    const unsigned short* wq = ws;                  // 196608 bf16
    const unsigned short* wp = ws + 48 * 8 * 512;   //  65536 bf16

    // 1) weight convert + swizzle (bf16 B-operand layout)
    const int total = (48 * 8 * 512) + (16 * 8 * 512);
    prep_weights_kernel<<<(total + 255) / 256, 256, 0, stream>>>(qkv_w, proj_w, ws);

    // 2) fused attention, one block per window
    win_attn_kernel<<<2048, 256, 0, stream>>>(x, mask, qkv_b, proj_b, bias_table,
                                              wq, wp, out);
}
